// NormalRefinement_20641612824905
// MI455X (gfx1250) — matrix operand order repaired and checked by reading, not compiled, not run on hardware
//
#include <hip/hip_runtime.h>

// ---------------- problem constants ----------------
constexpr int H1 = 481, W1 = 641;
constexpr int HW1 = H1 * W1;          // 308321
constexpr int H2 = 241, W2 = 321;
constexpr int HW2 = H2 * W2;          // 77361

// ---------------- types ----------------
typedef __attribute__((ext_vector_type(16))) __bf16        bf16x16;
typedef __attribute__((ext_vector_type(8)))  float         f32x8;
typedef __attribute__((ext_vector_type(8)))  unsigned short ushort8;

__device__ __forceinline__ unsigned short f2bf(float f) {
    unsigned int u = __float_as_uint(f);
    unsigned int r = u + 0x7FFFu + ((u >> 16) & 1u);   // round-to-nearest-even
    return (unsigned short)(r >> 16);
}
__device__ __forceinline__ float bf2f(unsigned short h) {
    return __uint_as_float(((unsigned int)h) << 16);
}

// ---------------- points = grid * 2^depth ----------------
__global__ void k_points(const float* __restrict__ grid, const float* __restrict__ up,
                         float* __restrict__ pts) {
    int p = blockIdx.x * blockDim.x + threadIdx.x;
    if (p >= HW1) return;
    float e = exp2f(up[p]);
    pts[(size_t)p * 3 + 0] = grid[p] * e;
    pts[(size_t)p * 3 + 1] = grid[HW1 + p] * e;
    pts[(size_t)p * 3 + 2] = grid[2 * HW1 + p] * e;
}

// ---------------- 81-tap gated plane fit ----------------
__global__ void k_planefit(const float* __restrict__ fc8n, const float* __restrict__ pts,
                           float* __restrict__ nrm, unsigned short* __restrict__ act0) {
    int p = blockIdx.x * blockDim.x + threadIdx.x;
    if (p >= HW1) return;
    int y = p / W1, x = p - y * W1;
    float c0 = fc8n[(size_t)p * 3], c1 = fc8n[(size_t)p * 3 + 1], c2 = fc8n[(size_t)p * 3 + 2];
    float a = 0.f, b = 0.f, cm = 0.f, d = 0.f, e = 0.f, f = 0.f;
    float t0 = 0.f, t1 = 0.f, t2 = 0.f;
    for (int i = 0; i < 9; ++i) {
        int yy = y + i * 4 - 16;
        if ((unsigned)yy >= (unsigned)H1) continue;   // zero padding -> angle 0 -> invalid
        for (int j = 0; j < 9; ++j) {
            int xx = x + j * 4 - 16;
            if ((unsigned)xx >= (unsigned)W1) continue;
            size_t q = ((size_t)yy * W1 + xx) * 3;
            float n0 = fc8n[q], n1 = fc8n[q + 1], n2 = fc8n[q + 2];
            float ang = n0 * c0 + n1 * c1 + n2 * c2;
            if (ang > 0.95f) {
                float a0 = pts[q], a1 = pts[q + 1], a2 = pts[q + 2];
                a += a0 * a0; b += a1 * a1; cm += a2 * a2;
                d += a0 * a1; e += a0 * a2; f += a1 * a2;
                t0 += a0; t1 += a1; t2 += a2;
            }
        }
    }
    float i00 = b * cm - f * f;
    float i01 = e * f - d * cm;
    float i02 = d * f - e * b;
    float det = a * i00 + d * i01 + e * i02;
    float g0, g1, g2;
    if (det > 1e-5f) {
        float i11 = a * cm - e * e;
        float i12 = e * d - a * f;
        float i22 = a * b - d * d;
        float inv = 1.0f / det;
        g0 = (i00 * t0 + i01 * t1 + i02 * t2) * inv;
        g1 = (i01 * t0 + i11 * t1 + i12 * t2) * inv;
        g2 = (i02 * t0 + i12 * t1 + i22 * t2) * inv;
    } else { g0 = t0; g1 = t1; g2 = t2; }
    float nn = fmaxf(sqrtf(g0 * g0 + g1 * g1 + g2 * g2), 1e-12f);
    float s = 10.0f / nn;
    g0 *= s; g1 *= s; g2 *= s;
    nrm[(size_t)p * 3 + 0] = g0;
    nrm[(size_t)p * 3 + 1] = g1;
    nrm[(size_t)p * 3 + 2] = g2;
    unsigned short* o = act0 + (size_t)p * 32;
    o[0] = f2bf(g0); o[1] = f2bf(g1); o[2] = f2bf(g2);
#pragma unroll
    for (int k = 3; k < 32; ++k) o[k] = 0;
}

// ---------------- weight fragment packing (OIHW f32 -> A-fragment bf16) ----------------
__global__ void k_packw(const float* __restrict__ w, unsigned short* __restrict__ wpk,
                        int Cout, int Cin, int KH, int KW, int nkc, int total) {
    int idx = blockIdx.x * blockDim.x + threadIdx.x;
    if (idx >= total) return;
    int e    = idx & 15;
    int lane = (idx >> 4) & 31;
    int blk  = idx >> 9;
    int cc   = blk % nkc; blk /= nkc;
    int ntap = KH * KW;
    int tap  = blk % ntap;
    int mt   = blk / ntap;
    int g = lane >> 4, m = lane & 15;
    int k = e + 16 * g;                          // contiguous k order: k = 16*(lane>>4) + e, matches B loads
    int cin = cc * 32 + k;
    int o = mt * 16 + m;
    float v = 0.f;
    if (o < Cout && cin < Cin)
        v = w[(((size_t)o * Cin + cin) * KH + tap / KW) * KW + (tap % KW)];
    wpk[idx] = f2bf(v);
}

__global__ void k_packb(const float* __restrict__ b, float* __restrict__ bp, int Cout, int Mpad) {
    int i = blockIdx.x * blockDim.x + threadIdx.x;
    if (i >= Mpad) return;
    bp[i] = (i < Cout) ? b[i] : 0.f;
}

// ---------------- implicit-GEMM conv ----------------
// one wave = 16 out-channels x 64 pixels (8x8 tile, 4 accumulators).
// interior tiles take an unguarded fast path (uniform scalar branch).
__global__ __launch_bounds__(256) void k_conv(
    const unsigned short* __restrict__ in, const unsigned short* __restrict__ wpk,
    const float* __restrict__ biasPad,
    unsigned short* __restrict__ obf, float* __restrict__ of32,
    int H, int W, int Cpad, int Mpad, int KH, int KW, int pad, int dil, int relu) {
    const int mtiles = Mpad >> 4;
    const int tilesX = (W + 7) >> 3;
    const int tilesY = (H + 7) >> 3;
    int wave = blockIdx.x * (blockDim.x >> 5) + (threadIdx.x >> 5);
    if (wave >= mtiles * tilesX * tilesY) return;   // wave-uniform exit: EXEC all-ones for WMMA
    int lane = threadIdx.x & 31;
    int mt = wave % mtiles;
    int t2 = wave / mtiles;
    int bx = t2 % tilesX, by = t2 / tilesX;
    int g = lane >> 4, n = lane & 15;
    int px  = (bx << 3) + (n & 7);
    int py0 = (by << 3) + (n >> 3);                 // subtile st -> row py0 + 2*st
    const int nkc  = Cpad >> 5;
    const int ntap = KH * KW;
    const long rowStride = (long)W * Cpad;          // elements per image row
    f32x8 a0 = {}, a1 = {}, a2 = {}, a3 = {};
    const unsigned short* wbase = wpk + (size_t)mt * ntap * nkc * 512 + lane * 16;

    // uniform interior test: every pixel of the 8x8 tile in-bounds for every tap
    bool interior = ((bx << 3) >= pad) && ((bx << 3) + 7 + (KW - 1) * dil - pad <= W - 1)
                 && ((by << 3) >= pad) && ((by << 3) + 7 + (KH - 1) * dil - pad <= H - 1);

    if (interior) {
        for (int tap = 0; tap < ntap; ++tap) {
            int ky = tap / KW, kx = tap - ky * KW;
            long base = ((long)(py0 + ky * dil - pad) * W + (px + kx * dil - pad)) * Cpad + g * 16;
            const unsigned short* aptr = wbase + (size_t)tap * nkc * 512;
            for (int cc = 0; cc < nkc; ++cc) {
                bf16x16 A = *(const bf16x16*)(aptr + (size_t)cc * 512);
                const unsigned short* bp0 = in + base + cc * 32;
                bf16x16 B0 = *(const bf16x16*)(bp0);
                bf16x16 B1 = *(const bf16x16*)(bp0 + 2 * rowStride);
                bf16x16 B2 = *(const bf16x16*)(bp0 + 4 * rowStride);
                bf16x16 B3 = *(const bf16x16*)(bp0 + 6 * rowStride);
                a0 = __builtin_amdgcn_wmma_f32_16x16x32_bf16(false, A, false, B0, (short)0, a0, false, false);
                a1 = __builtin_amdgcn_wmma_f32_16x16x32_bf16(false, A, false, B1, (short)0, a1, false, false);
                a2 = __builtin_amdgcn_wmma_f32_16x16x32_bf16(false, A, false, B2, (short)0, a2, false, false);
                a3 = __builtin_amdgcn_wmma_f32_16x16x32_bf16(false, A, false, B3, (short)0, a3, false, false);
            }
        }
    } else {
        for (int tap = 0; tap < ntap; ++tap) {
            int ky = tap / KW, kx = tap - ky * KW;
            int yy0 = py0 + ky * dil - pad;
            int xx  = px  + kx * dil - pad;
            bool okx = (unsigned)xx < (unsigned)W;
            bool ok0 = okx && (unsigned)(yy0)     < (unsigned)H;
            bool ok1 = okx && (unsigned)(yy0 + 2) < (unsigned)H;
            bool ok2 = okx && (unsigned)(yy0 + 4) < (unsigned)H;
            bool ok3 = okx && (unsigned)(yy0 + 6) < (unsigned)H;
            long base = ((long)yy0 * W + xx) * Cpad + g * 16;
            const unsigned short* aptr = wbase + (size_t)tap * nkc * 512;
            for (int cc = 0; cc < nkc; ++cc) {
                bf16x16 A = *(const bf16x16*)(aptr + (size_t)cc * 512);
                bf16x16 B0 = {}, B1 = {}, B2 = {}, B3 = {};
                if (ok0) B0 = *(const bf16x16*)(in + base + cc * 32);
                if (ok1) B1 = *(const bf16x16*)(in + base + 2 * rowStride + cc * 32);
                if (ok2) B2 = *(const bf16x16*)(in + base + 4 * rowStride + cc * 32);
                if (ok3) B3 = *(const bf16x16*)(in + base + 6 * rowStride + cc * 32);
                a0 = __builtin_amdgcn_wmma_f32_16x16x32_bf16(false, A, false, B0, (short)0, a0, false, false);
                a1 = __builtin_amdgcn_wmma_f32_16x16x32_bf16(false, A, false, B1, (short)0, a1, false, false);
                a2 = __builtin_amdgcn_wmma_f32_16x16x32_bf16(false, A, false, B2, (short)0, a2, false, false);
                a3 = __builtin_amdgcn_wmma_f32_16x16x32_bf16(false, A, false, B3, (short)0, a3, false, false);
            }
        }
    }

    // epilogue: D layout -> M = 8*(lane>>4)+reg, N = lane&15
    int mBase = mt * 16 + g * 8;
    const float* bpD = biasPad + mBase;
    float bb[8];
#pragma unroll
    for (int r = 0; r < 8; ++r) bb[r] = bpD[r];

#define STORE_ST(STI, ACC)                                                     \
    {                                                                          \
        int yy = py0 + 2 * (STI);                                              \
        if (yy < H && px < W) {                                                \
            size_t ob = ((size_t)yy * W + px) * Mpad + mBase;                  \
            if (of32) {                                                        \
                float* o = of32 + ob;                                          \
                _Pragma("unroll") for (int r = 0; r < 8; ++r) {                \
                    float v = (ACC)[r] + bb[r];                                \
                    if (relu) v = fmaxf(v, 0.f);                               \
                    o[r] = v;                                                  \
                }                                                              \
            } else {                                                           \
                ushort8 st;                                                    \
                _Pragma("unroll") for (int r = 0; r < 8; ++r) {                \
                    float v = (ACC)[r] + bb[r];                                \
                    if (relu) v = fmaxf(v, 0.f);                               \
                    st[r] = f2bf(v);                                           \
                }                                                              \
                *(ushort8*)(obf + ob) = st;                                    \
            }                                                                  \
        }                                                                      \
    }
    STORE_ST(0, a0)
    STORE_ST(1, a1)
    STORE_ST(2, a2)
    STORE_ST(3, a3)
#undef STORE_ST
}

// ---------------- 3x3 s2 p1 maxpool (HWC64 bf16) ----------------
__global__ void k_pool(const unsigned short* __restrict__ in, unsigned short* __restrict__ out) {
    int idx = blockIdx.x * blockDim.x + threadIdx.x;
    if (idx >= HW2 * 64) return;
    int c = idx & 63;
    int pix = idx >> 6;
    int ox = pix % W2, oy = pix / W2;
    float m = -3.402823466e38f;
    for (int dy = 0; dy < 3; ++dy) {
        int y = 2 * oy - 1 + dy;
        if ((unsigned)y >= (unsigned)H1) continue;
        for (int dx = 0; dx < 3; ++dx) {
            int x = 2 * ox - 1 + dx;
            if ((unsigned)x >= (unsigned)W1) continue;
            m = fmaxf(m, bf2f(in[((size_t)y * W1 + x) * 64 + c]));
        }
    }
    out[(size_t)pix * 64 + c] = f2bf(m);
}

// ---------------- resize(bilinear,ac) + sum + normalize + concat -> HWC32 bf16 ----------------
__global__ void k_concat(const float* __restrict__ fc8n, const float* __restrict__ inputs,
                         const float* __restrict__ nrm, const float* __restrict__ c9,
                         unsigned short* __restrict__ out) {
    int p = blockIdx.x * blockDim.x + threadIdx.x;
    if (p >= HW1) return;
    int y = p / W1, x = p - y * W1;
    int y0 = y >> 1; float wy = (y & 1) ? 0.5f : 0.f; int y1 = (y0 + 1 < H2) ? y0 + 1 : H2 - 1;
    int x0 = x >> 1; float wx = (x & 1) ? 0.5f : 0.f; int x1 = (x0 + 1 < W2) ? x0 + 1 : W2 - 1;
    size_t b00 = ((size_t)y0 * W2 + x0) * 16, b01 = ((size_t)y0 * W2 + x1) * 16;
    size_t b10 = ((size_t)y1 * W2 + x0) * 16, b11 = ((size_t)y1 * W2 + x1) * 16;
    float s[3];
#pragma unroll
    for (int c = 0; c < 3; ++c) {
        float v00 = c9[b00 + c], v01 = c9[b01 + c], v10 = c9[b10 + c], v11 = c9[b11 + c];
        float noise = (v00 * (1.f - wy) + v10 * wy) * (1.f - wx)
                    + (v01 * (1.f - wy) + v11 * wy) * wx;
        s[c] = nrm[(size_t)p * 3 + c] + noise;
    }
    float nn = fmaxf(sqrtf(s[0] * s[0] + s[1] * s[1] + s[2] * s[2]), 1e-12f);
    float inv = 1.0f / nn;
    unsigned short* o = out + (size_t)p * 32;
    o[0] = f2bf(fc8n[(size_t)p * 3 + 0]);
    o[1] = f2bf(fc8n[(size_t)p * 3 + 1]);
    o[2] = f2bf(fc8n[(size_t)p * 3 + 2]);
    o[3] = f2bf(s[0] * inv);
    o[4] = f2bf(s[1] * inv);
    o[5] = f2bf(s[2] * inv);
    o[6] = f2bf(inputs[p] * (1.f / 255.f));
    o[7] = f2bf(inputs[HW1 + p] * (1.f / 255.f));
    o[8] = f2bf(inputs[2 * HW1 + p] * (1.f / 255.f));
#pragma unroll
    for (int k = 9; k < 32; ++k) o[k] = 0;
}

// ---------------- final channel normalize -> d_out (HWC3 f32) ----------------
__global__ void k_outnorm(const float* __restrict__ yf, float* __restrict__ out) {
    int p = blockIdx.x * blockDim.x + threadIdx.x;
    if (p >= HW1) return;
    float v0 = yf[(size_t)p * 16 + 0];
    float v1 = yf[(size_t)p * 16 + 1];
    float v2 = yf[(size_t)p * 16 + 2];
    float nn = fmaxf(sqrtf(v0 * v0 + v1 * v1 + v2 * v2), 1e-12f);
    float inv = 1.0f / nn;
    out[(size_t)p * 3 + 0] = v0 * inv;
    out[(size_t)p * 3 + 1] = v1 * inv;
    out[(size_t)p * 3 + 2] = v2 * inv;
}

// ---------------- host orchestration ----------------
extern "C" void kernel_launch(void* const* d_in, const int* in_sizes, int n_in,
                              void* d_out, int out_size, void* d_ws, size_t ws_size,
                              hipStream_t stream) {
    (void)in_sizes; (void)n_in; (void)out_size; (void)ws_size;
    const float* fc8n   = (const float*)d_in[0];
    const float* up     = (const float*)d_in[1];
    const float* grid   = (const float*)d_in[2];
    const float* inputs = (const float*)d_in[3];
    const float *cw[9], *cb[9], *fw[7], *fb[7];
    for (int i = 0; i < 9; ++i) { cw[i] = (const float*)d_in[4 + 2 * i];  cb[i] = (const float*)d_in[5 + 2 * i]; }
    for (int i = 0; i < 7; ++i) { fw[i] = (const float*)d_in[22 + 2 * i]; fb[i] = (const float*)d_in[23 + 2 * i]; }

    char* ws = (char*)d_ws;
    size_t off = 0;
    auto alloc = [&](size_t bytes) -> char* {
        char* p = ws + off;
        off += (bytes + 255) & ~(size_t)255;
        return p;
    };
    float* R_norm = (float*)alloc((size_t)HW1 * 3 * 4);
    size_t smallSz = (size_t)HW2 * 16 * 4;                             // conv9 out (f32, 16ch)
    if ((size_t)HW1 * 3 * 4 > smallSz) smallSz = (size_t)HW1 * 3 * 4;  // also holds points
    char* R_small = alloc(smallSz);
    size_t poolSz = (size_t)HW2 * 512 * 2;                             // max activation
    if ((size_t)HW1 * 128 * 2 > poolSz) poolSz = (size_t)HW1 * 128 * 2;
    unsigned short* P0 = (unsigned short*)alloc(poolSz);
    unsigned short* P1 = (unsigned short*)alloc(poolSz);

    struct CD { const float* w; const float* b; int Cin, Cout, KH, KW; };
    const CD cds[16] = {
        {cw[0], cb[0],   3,  64, 3, 3}, {cw[1], cb[1],  64,  64, 3, 3},
        {cw[2], cb[2],  64, 128, 3, 3}, {cw[3], cb[3], 128, 128, 3, 3},
        {cw[4], cb[4], 128, 256, 3, 3}, {cw[5], cb[5], 256, 256, 3, 3},
        {cw[6], cb[6], 256, 256, 3, 3}, {cw[7], cb[7], 256, 512, 1, 1},
        {cw[8], cb[8], 512,   3, 3, 3},
        {fw[0], fb[0],   9, 128, 3, 3}, {fw[1], fb[1], 128, 128, 3, 3},
        {fw[2], fb[2], 128, 128, 3, 3}, {fw[3], fb[3], 128, 128, 3, 3},
        {fw[4], fb[4], 128, 128, 3, 3}, {fw[5], fb[5], 128, 128, 3, 3},
        {fw[6], fb[6], 128,   3, 3, 3}};

    unsigned short* wpk[16];
    float* bpd[16];
    int Cpad[16], Mpad[16];
    for (int i = 0; i < 16; ++i) {
        Cpad[i] = ((cds[i].Cin + 31) / 32) * 32;
        Mpad[i] = ((cds[i].Cout + 15) / 16) * 16;
        int nkc = Cpad[i] / 32;
        int total = (Mpad[i] / 16) * cds[i].KH * cds[i].KW * nkc * 512;
        wpk[i] = (unsigned short*)alloc((size_t)total * 2);
        bpd[i] = (float*)alloc((size_t)Mpad[i] * 4);
        k_packw<<<(total + 255) / 256, 256, 0, stream>>>(
            cds[i].w, wpk[i], cds[i].Cout, cds[i].Cin, cds[i].KH, cds[i].KW, nkc, total);
        k_packb<<<(Mpad[i] + 63) / 64, 64, 0, stream>>>(cds[i].b, bpd[i], cds[i].Cout, Mpad[i]);
    }

    auto conv = [&](int i, const unsigned short* in, unsigned short* obf, float* of32,
                    int H, int W, int pad, int dil, int relu) {
        long waves = (long)(Mpad[i] / 16) * ((W + 7) / 8) * ((H + 7) / 8);
        int blocks = (int)((waves + 7) / 8);
        k_conv<<<blocks, 256, 0, stream>>>(in, wpk[i], bpd[i], obf, of32,
                                           H, W, Cpad[i], Mpad[i],
                                           cds[i].KH, cds[i].KW, pad, dil, relu);
    };

    const int PB = 256;
    const int GB1 = (HW1 + PB - 1) / PB;

    // stage A: points + plane fit (writes R_norm f32, act0 bf16 into P0)
    float* Pts = (float*)R_small;
    k_points<<<GB1, PB, 0, stream>>>(grid, up, Pts);
    k_planefit<<<GB1, PB, 0, stream>>>(fc8n, Pts, R_norm, P0);

    // cnn_refine
    conv(0, P0, P1, nullptr, H1, W1, 1, 1, 1);            // 3 -> 64
    conv(1, P1, P0, nullptr, H1, W1, 1, 1, 1);            // 64 -> 64
    k_pool<<<(HW2 * 64 + PB - 1) / PB, PB, 0, stream>>>(P0, P1);
    conv(2, P1, P0, nullptr, H2, W2, 1, 1, 1);            // 64 -> 128
    conv(3, P0, P1, nullptr, H2, W2, 1, 1, 1);            // 128 -> 128
    conv(4, P1, P0, nullptr, H2, W2, 1, 1, 1);            // 128 -> 256
    conv(5, P0, P1, nullptr, H2, W2, 1, 1, 1);            // 256 -> 256
    conv(6, P1, P0, nullptr, H2, W2, 1, 1, 1);            // 256 -> 256
    conv(7, P0, P1, nullptr, H2, W2, 0, 1, 1);            // 1x1, 256 -> 512
    float* C9 = (float*)R_small;                          // points now dead
    conv(8, P1, nullptr, C9, H2, W2, 1, 1, 0);            // 512 -> 3 (f32 out)

    // stage C: resize + sum + normalize + concat -> 9ch (padded 32) bf16
    k_concat<<<GB1, PB, 0, stream>>>(fc8n, inputs, R_norm, C9, P0);

    // final stack
    conv( 9, P0, P1, nullptr, H1, W1, 2, 2, 1);           // 9 -> 128, dil2
    conv(10, P1, P0, nullptr, H1, W1, 2, 2, 1);
    conv(11, P0, P1, nullptr, H1, W1, 2, 2, 1);
    conv(12, P1, P0, nullptr, H1, W1, 1, 1, 1);
    conv(13, P0, P1, nullptr, H1, W1, 1, 1, 1);
    conv(14, P1, P0, nullptr, H1, W1, 1, 1, 1);
    float* YF = (float*)P1;
    conv(15, P0, nullptr, YF, H1, W1, 1, 1, 0);           // 128 -> 3 (f32 out)

    k_outnorm<<<GB1, PB, 0, stream>>>(YF, (float*)d_out);
}
